// SemanticModule_27650999452286
// MI455X (gfx1250) — matrix operand order repaired and checked by reading, not compiled, not run on hardware
//
#include <hip/hip_runtime.h>
#include <hip/hip_bf16.h>
#include <stdint.h>

// Problem constants (match reference)
constexpr int B_ = 4, S_ = 2048, L_ = 512, N_ = 8192;
constexpr int D_ = 1024;          // 2*L
constexpr int T_ = B_ * S_;       // 8192 tokens
constexpr float GBS = 0.8f;       // GRAPH_BIAS_SCALE
constexpr float CC  = 0.01f;      // COMMITMENT_COST

typedef __attribute__((ext_vector_type(16))) __bf16 v16bf;
typedef __attribute__((ext_vector_type(8)))  __bf16 v8bf;
typedef __attribute__((ext_vector_type(8)))  float  v8f;

// ---------------------------------------------------------------------------
// Prep 1: z = concat(real, imag) -> bf16 hi/lo rows + ||z||^2 ; zero loss acc
// ---------------------------------------------------------------------------
__global__ __launch_bounds__(256) void prep_z(
    const float* __restrict__ gr, const float* __restrict__ gi,
    __bf16* __restrict__ zh, __bf16* __restrict__ zl,
    float* __restrict__ znorm, float* __restrict__ loss_acc)
{
    const int tok = blockIdx.x, tid = threadIdx.x;
    if (tok == 0 && tid == 0) *loss_acc = 0.f;
    float ss = 0.f;
#pragma unroll
    for (int i = 0; i < D_ / 256; ++i) {
        const int d = tid + i * 256;
        const float x = (d < L_) ? gr[(size_t)tok * L_ + d]
                                 : gi[(size_t)tok * L_ + (d - L_)];
        const __bf16 h = (__bf16)x;               // RNE
        const __bf16 l = (__bf16)(x - (float)h);  // residual
        zh[(size_t)tok * D_ + d] = h;
        zl[(size_t)tok * D_ + d] = l;
        ss += x * x;
    }
    __shared__ float red[256];
    red[tid] = ss; __syncthreads();
    for (int o = 128; o > 0; o >>= 1) { if (tid < o) red[tid] += red[tid + o]; __syncthreads(); }
    if (tid == 0) znorm[tok] = red[0];
}

// ---------------------------------------------------------------------------
// Prep 2: codebook -> transposed bf16 hi/lo  cbT[k][n]  (LDS-tiled transpose)
// ---------------------------------------------------------------------------
__global__ __launch_bounds__(256) void prep_cbT(
    const float* __restrict__ cb, __bf16* __restrict__ th, __bf16* __restrict__ tl)
{
    __shared__ float tile[32][33];
    const int k0 = blockIdx.x * 32, n0 = blockIdx.y * 32;
    const int tx = threadIdx.x, ty = threadIdx.y;
    for (int i = ty; i < 32; i += 8)
        tile[i][tx] = cb[(size_t)(n0 + i) * D_ + k0 + tx];
    __syncthreads();
    for (int i = ty; i < 32; i += 8) {
        const float x = tile[tx][i];                 // = cb[n0+tx][k0+i]
        const __bf16 h = (__bf16)x;
        const __bf16 l = (__bf16)(x - (float)h);
        const size_t o = (size_t)(k0 + i) * N_ + n0 + tx;  // coalesced over tx
        th[o] = h; tl[o] = l;
    }
}

// ---------------------------------------------------------------------------
// Prep 3: ||codebook_n||^2
// ---------------------------------------------------------------------------
__global__ __launch_bounds__(256) void prep_cnorm(
    const float* __restrict__ cb, float* __restrict__ cn)
{
    const int n = blockIdx.x, tid = threadIdx.x;
    float ss = 0.f;
#pragma unroll
    for (int i = 0; i < D_ / 256; ++i) {
        const float x = cb[(size_t)n * D_ + tid + i * 256];
        ss += x * x;
    }
    __shared__ float red[256];
    red[tid] = ss; __syncthreads();
    for (int o = 128; o > 0; o >>= 1) { if (tid < o) red[tid] += red[tid + o]; __syncthreads(); }
    if (tid == 0) cn[n] = red[0];
}

// ---------------------------------------------------------------------------
// Main fused kernel: WMMA GEMM (hi/lo bf16 split) + async-prefetched graph
// bias + argmin.  WG = 8 waves = 32 tokens (2 M-tiles per wave).
// Each wave: 4 n-tile accumulators x 2 M-tiles = 24 WMMA per k-step (8 b128 A loads, 128 2-byte B loads).
// Adjacency bias tile (32x64 f32) is async-copied into per-wave LDS while
// the k-loop runs (global_load_async_to_lds_b128 / s_wait_asynccnt).
// ---------------------------------------------------------------------------
__global__ __launch_bounds__(256) void vq_argmin(
    const __bf16* __restrict__ zh, const __bf16* __restrict__ zl,
    const __bf16* __restrict__ cth, const __bf16* __restrict__ ctl,
    const float* __restrict__ znorm, const float* __restrict__ cnorm,
    const int* __restrict__ prev, const float* __restrict__ adj,
    float* __restrict__ minval, int* __restrict__ minidx)
{
    __shared__ float s_adj[8][32][64];   // 64 KB: per-wave bias tile; aliased for reduce
    __shared__ float s_zn[32];
    __shared__ int   s_pv[32];

    const int tid  = threadIdx.x;
    const int lane = tid & 31;
    const int wv   = tid >> 5;
    const int tok0 = blockIdx.x * 32;

    if (tid < 32) { s_zn[tid] = znorm[tok0 + tid]; s_pv[tid] = prev[tok0 + tid]; }
    __syncthreads();

    // A-operand addressing (16-bit A 16x32 layout: lane half selects K-octet)
    const int am = lane & 15, ah = lane >> 4;
    const __bf16* zrh0 = zh + (size_t)(tok0 + am) * D_;
    const __bf16* zrl0 = zl + (size_t)(tok0 + am) * D_;
    const __bf16* zrh1 = zrh0 + (size_t)16 * D_;
    const __bf16* zrl1 = zrl0 + (size_t)16 * D_;

    // async copy lane mapping: 2 rows per instruction, 16 lanes x 16B per row
    const int      pmh          = lane >> 4;            // row parity within pair
    const int      pnf          = (lane & 15) * 4;      // float offset in 64-wide row
    const unsigned lds_adj_base = (unsigned)(size_t)&s_adj[wv][0][0];

    float vmin[2][8]; int vidx[2][8];
#pragma unroll
    for (int mt = 0; mt < 2; ++mt)
#pragma unroll
        for (int v = 0; v < 8; ++v) { vmin[mt][v] = 3.4e38f; vidx[mt][v] = 0; }

    for (int blk = 0; blk < 16; ++blk) {
        const int n0 = (blk * 8 + wv) * 64;   // this wave's 64-column block

        // ---- issue async prefetch of adjacency bias tile for this block ----
        // rows m = 0..31 (tokens), cols n0..n0+63 ; 8 KB into s_adj[wv]
#pragma unroll
        for (int i = 0; i < 16; ++i) {
            const int m = 2 * i + pmh;
            const float* g = adj + (size_t)s_pv[m] * N_ + (size_t)n0 + pnf;
            const unsigned lo = lds_adj_base + (unsigned)(m * 64 + pnf) * 4u;
            asm volatile("global_load_async_to_lds_b128 %0, %1, off"
                         :: "v"(lo), "v"(g) : "memory");
        }

        // ---- GEMM k-loop (overlaps with async copies) ----
        v8f acc[2][4];
#pragma unroll
        for (int mt = 0; mt < 2; ++mt)
#pragma unroll
            for (int t = 0; t < 4; ++t) {
                v8f zq = {0.f, 0.f, 0.f, 0.f, 0.f, 0.f, 0.f, 0.f};
                acc[mt][t] = zq;
            }

        for (int kk = 0; kk < 32; ++kk) {
            const int k0 = kk * 32;
            v16bf a_h0, a_l0, a_h1, a_l1;
            {
                const v8bf p0 = *(const v8bf*)(zrh0 + k0 + ah * 8);
                const v8bf p1 = *(const v8bf*)(zrh0 + k0 + 16 + ah * 8);
                const v8bf q0 = *(const v8bf*)(zrl0 + k0 + ah * 8);
                const v8bf q1 = *(const v8bf*)(zrl0 + k0 + 16 + ah * 8);
                const v8bf p2 = *(const v8bf*)(zrh1 + k0 + ah * 8);
                const v8bf p3 = *(const v8bf*)(zrh1 + k0 + 16 + ah * 8);
                const v8bf q2 = *(const v8bf*)(zrl1 + k0 + ah * 8);
                const v8bf q3 = *(const v8bf*)(zrl1 + k0 + 16 + ah * 8);
#pragma unroll
                for (int e = 0; e < 8; ++e) {
                    a_h0[e] = p0[e]; a_h0[e + 8] = p1[e];
                    a_l0[e] = q0[e]; a_l0[e + 8] = q1[e];
                    a_h1[e] = p2[e]; a_h1[e + 8] = p3[e];
                    a_l1[e] = q2[e]; a_l1[e + 8] = q3[e];
                }
            }
            const size_t rb = (size_t)(k0 + (lane >> 4) * 8) * N_;
#pragma unroll
            for (int t = 0; t < 4; ++t) {
                const size_t o = rb + (size_t)(n0 + 16 * t + (lane & 15));
                v16bf b_h, b_l;
#pragma unroll
                for (int e = 0; e < 8; ++e) {
                    b_h[e] = cth[o + (size_t)e * N_]; b_h[e + 8] = cth[o + (size_t)(16 + e) * N_];
                    b_l[e] = ctl[o + (size_t)e * N_]; b_l[e + 8] = ctl[o + (size_t)(16 + e) * N_];
                }
                // dot += zh*ch + zh*cl + zl*ch   (drop lo*lo)
                acc[0][t] = __builtin_amdgcn_wmma_f32_16x16x32_bf16(
                                false, a_h0, false, b_h, (short)0, acc[0][t], false, false);
                acc[0][t] = __builtin_amdgcn_wmma_f32_16x16x32_bf16(
                                false, a_h0, false, b_l, (short)0, acc[0][t], false, false);
                acc[0][t] = __builtin_amdgcn_wmma_f32_16x16x32_bf16(
                                false, a_l0, false, b_h, (short)0, acc[0][t], false, false);
                acc[1][t] = __builtin_amdgcn_wmma_f32_16x16x32_bf16(
                                false, a_h1, false, b_h, (short)0, acc[1][t], false, false);
                acc[1][t] = __builtin_amdgcn_wmma_f32_16x16x32_bf16(
                                false, a_h1, false, b_l, (short)0, acc[1][t], false, false);
                acc[1][t] = __builtin_amdgcn_wmma_f32_16x16x32_bf16(
                                false, a_l1, false, b_h, (short)0, acc[1][t], false, false);
            }
        }

        // ---- bias tile must be resident now ----
        asm volatile("s_wait_asynccnt 0x0" ::: "memory");

        // Distances + graph bias + running argmin.
        // C layout: lane col = lane&15, row m = v + 8*(lane>>4)
        const int nl = lane & 15, mh = (lane >> 4) * 8;
#pragma unroll
        for (int t = 0; t < 4; ++t) {
            const int n = n0 + 16 * t + nl;
            const float cn = cnorm[n];
#pragma unroll
            for (int mt = 0; mt < 2; ++mt)
#pragma unroll
                for (int v = 0; v < 8; ++v) {
                    const int m = mt * 16 + mh + v;
                    const float g   = s_adj[wv][m][16 * t + nl];
                    const float sig = 1.f / (1.f + __expf(-g));
                    const float dst = s_zn[m] + cn - 2.f * acc[mt][t][v] - GBS * sig;
                    if (dst < vmin[mt][v]) { vmin[mt][v] = dst; vidx[mt][v] = n; }
                }
        }
    }

    // ---- cross-lane / cross-wave argmin reduction (alias s_adj as scratch) ----
    __syncthreads();                       // all waves done reading s_adj
    float* s_mv = (float*)&s_adj[0][0][0]; // 4096 floats (16 KB)
    int*   s_mi = (int*)  &s_adj[2][0][0]; // 4096 ints   (16 KB)
#pragma unroll
    for (int mt = 0; mt < 2; ++mt)
#pragma unroll
        for (int v = 0; v < 8; ++v) {
            const int slot = ((wv * 2 + mt) * 32 + lane) * 8 + v;
            s_mv[slot] = vmin[mt][v]; s_mi[slot] = vidx[mt][v];
        }
    __syncthreads();
    if (tid < 32) {
        const int m = tid, mt = m >> 4, mm = m & 15;
        float bv = 3.4e38f; int bi = 0x7fffffff;
        for (int w = 0; w < 8; ++w)
            for (int j = 0; j < 16; ++j) {
                const int l = (mm >> 3) * 16 + j;
                const int slot = ((w * 2 + mt) * 32 + l) * 8 + (mm & 7);
                const float x = s_mv[slot]; const int ix = s_mi[slot];
                if (x < bv || (x == bv && ix < bi)) { bv = x; bi = ix; }
            }
        minval[tok0 + m] = bv;
        minidx[tok0 + m] = bi;
    }
}

// ---------------------------------------------------------------------------
// Epilogue: gather z_q, write proposals, salience/confidence heads, loss acc
// ---------------------------------------------------------------------------
__global__ __launch_bounds__(256) void epilogue(
    const float* __restrict__ gr, const float* __restrict__ gi,
    const float* __restrict__ cb,
    const float* __restrict__ salw, const float* __restrict__ salb,
    const float* __restrict__ confw, const float* __restrict__ confb,
    const float* __restrict__ minval, const int* __restrict__ minidx,
    float* __restrict__ out_r, float* __restrict__ out_i,
    float* __restrict__ out_s, float* __restrict__ out_c,
    float* __restrict__ out_mi, float* __restrict__ loss_acc)
{
    const int tok = blockIdx.x, tid = threadIdx.x;
    const int idx = minidx[tok];
    const float* crow = cb + (size_t)idx * D_;
    float lsum = 0.f, ssum = 0.f, csum = 0.f;
#pragma unroll
    for (int i = 0; i < D_ / 256; ++i) {
        const int d = tid + i * 256;
        const float q = crow[d];
        const float x = (d < L_) ? gr[(size_t)tok * L_ + d]
                                 : gi[(size_t)tok * L_ + (d - L_)];
        const float df = q - x;
        lsum += df * df;
        ssum += q * salw[d];
        csum += q * confw[d];
        if (d < L_) out_r[(size_t)tok * L_ + d] = q;
        else        out_i[(size_t)tok * L_ + (d - L_)] = q;
    }
    __shared__ float r1[256], r2[256], r3[256];
    r1[tid] = lsum; r2[tid] = ssum; r3[tid] = csum; __syncthreads();
    for (int o = 128; o > 0; o >>= 1) {
        if (tid < o) { r1[tid] += r1[tid + o]; r2[tid] += r2[tid + o]; r3[tid] += r3[tid + o]; }
        __syncthreads();
    }
    if (tid == 0) {
        atomicAdd(loss_acc, r1[0]);
        out_s[tok] = r2[0] + salb[0] + 0.1f * (-minval[tok]);
        const float cz = r3[0] + confb[0];
        out_c[tok] = 1.f / (1.f + __expf(-cz));
        out_mi[tok] = (float)idx;
    }
}

__global__ void finalize(const float* __restrict__ loss_acc, float* __restrict__ out_loss)
{
    *out_loss = (1.0f + CC) * (*loss_acc) / (float)((size_t)T_ * D_);
}

// ---------------------------------------------------------------------------
// Launch
// ---------------------------------------------------------------------------
extern "C" void kernel_launch(void* const* d_in, const int* in_sizes, int n_in,
                              void* d_out, int out_size, void* d_ws, size_t ws_size,
                              hipStream_t stream)
{
    const float* gw_real = (const float*)d_in[0];
    const float* gw_imag = (const float*)d_in[1];
    const int*   prev    = (const int*)  d_in[2];
    const float* codebook= (const float*)d_in[3];
    const float* adj     = (const float*)d_in[4];
    const float* sal_w   = (const float*)d_in[5];
    const float* sal_b   = (const float*)d_in[6];
    const float* conf_w  = (const float*)d_in[7];
    const float* conf_b  = (const float*)d_in[8];

    // Workspace carve-up (~64.1 MB)
    char* w = (char*)d_ws;
    __bf16* cth   = (__bf16*)w; w += (size_t)N_ * D_ * sizeof(__bf16);
    __bf16* ctl   = (__bf16*)w; w += (size_t)N_ * D_ * sizeof(__bf16);
    __bf16* zh    = (__bf16*)w; w += (size_t)T_ * D_ * sizeof(__bf16);
    __bf16* zl    = (__bf16*)w; w += (size_t)T_ * D_ * sizeof(__bf16);
    float*  cnorm = (float*)w;  w += (size_t)N_ * sizeof(float);
    float*  znorm = (float*)w;  w += (size_t)T_ * sizeof(float);
    float*  minv  = (float*)w;  w += (size_t)T_ * sizeof(float);
    int*    mini  = (int*)w;    w += (size_t)T_ * sizeof(int);
    float*  lacc  = (float*)w;  w += sizeof(float);

    // Output layout (flat concat, all float)
    float* out = (float*)d_out;
    float* out_r  = out;                       // B*S*L
    float* out_i  = out_r + (size_t)T_ * L_;   // B*S*L
    float* out_s  = out_i + (size_t)T_ * L_;   // B*S
    float* out_c  = out_s + T_;                // B*S
    float* out_l  = out_c + T_;                // 1
    float* out_mi = out_l + 1;                 // B*S

    prep_z    <<<T_, 256, 0, stream>>>(gw_real, gw_imag, zh, zl, znorm, lacc);
    prep_cnorm<<<N_, 256, 0, stream>>>(codebook, cnorm);
    prep_cbT  <<<dim3(D_ / 32, N_ / 32), dim3(32, 8), 0, stream>>>(codebook, cth, ctl);

    vq_argmin<<<T_ / 32, 256, 0, stream>>>(zh, zl, cth, ctl, znorm, cnorm,
                                           prev, adj, minv, mini);

    epilogue<<<T_, 256, 0, stream>>>(gw_real, gw_imag, codebook,
                                     sal_w, sal_b, conf_w, conf_b,
                                     minv, mini,
                                     out_r, out_i, out_s, out_c, out_mi, lacc);
    finalize<<<1, 1, 0, stream>>>(lacc, out_l);
}
